// XNOR_Net_FeedForwardNet_28200755265763
// MI455X (gfx1250) — 8-bit matrix operand order repaired and checked by reading, compiled, not run on hardware
//
#include <hip/hip_runtime.h>
#include <cstdint>
#include <cstddef>

#define N_SAMP 32768
#define D_IN   512
#define D_HID  2048
#define D_OUT  512

#define BM 128
#define BN 128
#define BK 128
#define LSTRIDE (BK + 16)
#define TILE_BYTES (BM * LSTRIDE)

typedef int          v16i  __attribute__((ext_vector_type(16)));
typedef float        v8f   __attribute__((ext_vector_type(8)));
typedef unsigned int u32x4 __attribute__((ext_vector_type(4)));
typedef int          i32x4 __attribute__((ext_vector_type(4)));
typedef int          i32x8 __attribute__((ext_vector_type(8)));

#if defined(__has_builtin)
#if __has_builtin(__builtin_amdgcn_tensor_load_to_lds) && __has_builtin(__builtin_amdgcn_s_wait_tensorcnt)
#define USE_TDM 1
#endif
#endif
#ifndef USE_TDM
#define USE_TDM 0
#endif

__device__ __forceinline__ unsigned char sign_fp8(float v) {
    return (v >= 0.f) ? (unsigned char)0x38 : (unsigned char)0xB8;
}

#if USE_TDM
__device__ __forceinline__ void tdm_load_tile(const unsigned char* gbase,
                                              unsigned lds_off, int K, int rows)
{
    const unsigned long long ga = (unsigned long long)(uintptr_t)gbase;

    u32x4 g0;
    g0[0] = 1u;
    g0[1] = lds_off;
    g0[2] = (unsigned)(ga & 0xFFFFFFFFu);
    g0[3] = (unsigned)((ga >> 32) & 0x01FFFFFFu)
          | (2u << 30);

    i32x8 g1;
    g1[0] = (1 << 20)
          | (4 << 22)
          | (3 << 25);
    g1[1] = (int)(((unsigned)K & 0xFFFFu) << 16);
    g1[2] = (int)(((unsigned)K >> 16) |
                  (((unsigned)rows & 0xFFFFu) << 16));
    g1[3] = (int)(((unsigned)rows >> 16) | (128u << 16));
    g1[4] = 128;
    g1[5] = K;
    g1[6] = 0;
    g1[7] = 0;

    i32x4 z4 = {0, 0, 0, 0};
#if __clang_major__ >= 23
    i32x8 z8 = {0, 0, 0, 0, 0, 0, 0, 0};
    __builtin_amdgcn_tensor_load_to_lds(g0, g1, z4, z4, z8, 0);
#else
    __builtin_amdgcn_tensor_load_to_lds(g0, g1, z4, z4, 0);
#endif
}
#endif

__global__ __launch_bounds__(256)
void prep_sign_scale(const float* __restrict__ in,
                     unsigned char* __restrict__ signs,
                     float* __restrict__ scale, int C)
{
    const int r = blockIdx.x;
    const float4* row  = (const float4*)(in + (size_t)r * C);
    uchar4*       srow = (uchar4*)(signs + (size_t)r * C);
    const int c4n = C >> 2;

    float s = 0.f;
    for (int c = threadIdx.x; c < c4n; c += blockDim.x) {
        float4 v = row[c];
        uchar4 o;
        o.x = sign_fp8(v.x); o.y = sign_fp8(v.y);
        o.z = sign_fp8(v.z); o.w = sign_fp8(v.w);
        srow[c] = o;
        s += fabsf(v.x) + fabsf(v.y) + fabsf(v.z) + fabsf(v.w);
    }
    for (int off = 16; off > 0; off >>= 1) s += __shfl_xor(s, off, 32);
    __shared__ float wsum[8];
    if ((threadIdx.x & 31) == 0) wsum[threadIdx.x >> 5] = s;
    __syncthreads();
    if (threadIdx.x == 0) {
        float tot = 0.f;
#pragma unroll
        for (int i = 0; i < 8; ++i) tot += wsum[i];
        scale[r] = tot / (float)C;
    }
}

__global__ __launch_bounds__(256)
void zero_f32(float* __restrict__ p, int n)
{
    int i = blockIdx.x * blockDim.x + threadIdx.x;
    if (i < n) p[i] = 0.f;
}

__global__ __launch_bounds__(256, 2)
void xnor_wmma_gemm(const unsigned char* __restrict__ Asign,
                    const unsigned char* __restrict__ Bsign,
                    const float* __restrict__ beta_in, float beta_scale,
                    const float* __restrict__ alpha,
                    const float* __restrict__ bias,
                    const float* __restrict__ prelu_w,
                    unsigned char* __restrict__ OutSign,
                    float* __restrict__ betasum_out,
                    float* __restrict__ OutF32,
                    int M, int NC, int K, int mode)
{
    __shared__ __align__(128) unsigned char lA[2][TILE_BYTES];
    __shared__ __align__(128) unsigned char lB[2][TILE_BYTES];
    __shared__ float rowsum[BM];

    const int t    = threadIdx.x;
    const int lane = t & 31;
    const int wave = t >> 5;
    const int wm   = wave & 1;
    const int wn   = wave >> 1;

    const int bm = blockIdx.x * BM;
    const int bn = blockIdx.y * BN;

    const int mrow = lane & 15;
    const int half = lane >> 4;

    v8f acc[4][2];
#pragma unroll
    for (int i = 0; i < 4; ++i)
#pragma unroll
        for (int j = 0; j < 2; ++j)
#pragma unroll
            for (int e = 0; e < 8; ++e) acc[i][j][e] = 0.f;

    const int ksteps = K / BK;

#if USE_TDM
    const unsigned ldsA[2] = { (unsigned)(uintptr_t)&lA[0][0],
                               (unsigned)(uintptr_t)&lA[1][0] };
    const unsigned ldsB[2] = { (unsigned)(uintptr_t)&lB[0][0],
                               (unsigned)(uintptr_t)&lB[1][0] };
    if (wave == 0) {
        tdm_load_tile(Asign + (size_t)bm * K, ldsA[0], K, M);
        tdm_load_tile(Bsign + (size_t)bn * K, ldsB[0], K, NC);
        __builtin_amdgcn_s_wait_tensorcnt(0);
    }
    __syncthreads();
#endif

    for (int ks = 0; ks < ksteps; ++ks) {
        const int cur = ks & 1;

#if USE_TDM
        if ((ks + 1) < ksteps && wave == 0) {
            const size_t kb = (size_t)(ks + 1) * BK;
            tdm_load_tile(Asign + (size_t)bm * K + kb, ldsA[cur ^ 1], K, M);
            tdm_load_tile(Bsign + (size_t)bn * K + kb, ldsB[cur ^ 1], K, NC);
        }
        const unsigned char* __restrict__ curA = lA[cur];
        const unsigned char* __restrict__ curB = lB[cur];
#else
        {
            const size_t kbase = (size_t)ks * BK;
#pragma unroll
            for (int it = 0; it < 4; ++it) {
                const int i   = t + it * 256;
                const int row = i >> 3;
                const int seg = i & 7;
                *(uint4*)&lA[0][row * LSTRIDE + seg * 16] =
                    *(const uint4*)(Asign + (size_t)(bm + row) * K + kbase + seg * 16);
                *(uint4*)&lB[0][row * LSTRIDE + seg * 16] =
                    *(const uint4*)(Bsign + (size_t)(bn + row) * K + kbase + seg * 16);
            }
            if (ks + 1 < ksteps) {
                const int prow = t >> 1;
                const int ph   = (t & 1) * 64;
                __builtin_prefetch(Asign + (size_t)(bm + prow) * K + kbase + BK + ph, 0, 1);
                __builtin_prefetch(Bsign + (size_t)(bn + prow) * K + kbase + BK + ph, 0, 1);
            }
        }
        __syncthreads();
        const unsigned char* __restrict__ curA = lA[0];
        const unsigned char* __restrict__ curB = lB[0];
#endif

        v16i bfrag[2];
#pragma unroll
        for (int tn = 0; tn < 2; ++tn) {
            const int cb = wn * 32 + tn * 16 + mrow;
#pragma unroll
            for (int g = 0; g < 4; ++g) {
                uint4 q = *(const uint4*)&curB[cb * LSTRIDE + g * 32 + half * 16];
                bfrag[tn][4 * g + 0] = (int)q.x;
                bfrag[tn][4 * g + 1] = (int)q.y;
                bfrag[tn][4 * g + 2] = (int)q.z;
                bfrag[tn][4 * g + 3] = (int)q.w;
            }
        }

#pragma unroll
        for (int tm = 0; tm < 4; ++tm) {
            const int rb = wm * 64 + tm * 16 + mrow;
            v16i afrag;
#pragma unroll
            for (int s = 0; s < 2; ++s)
#pragma unroll
                for (int p = 0; p < 4; ++p) {
                    uint2 q = *(const uint2*)&curA[rb * LSTRIDE + s * 64 + (p >> 1) * 32 + half * 16 + (p & 1) * 8];
                    afrag[s * 8 + p * 2 + 0] = (int)q.x;
                    afrag[s * 8 + p * 2 + 1] = (int)q.y;
                }
#pragma unroll
            for (int tn = 0; tn < 2; ++tn)
                acc[tm][tn] = __builtin_amdgcn_wmma_f32_16x16x128_fp8_fp8(
                    afrag, bfrag[tn], (short)0, acc[tm][tn], false, false);
        }

#if USE_TDM
        if (wave == 0 && (ks + 1) < ksteps)
            __builtin_amdgcn_s_wait_tensorcnt(0);
#endif
        __syncthreads();
    }

    const float pw = (mode == 0) ? prelu_w[0] : 0.f;

    if (mode == 0) {
        for (int i = t; i < BM; i += 256) rowsum[i] = 0.f;
        __syncthreads();
    }

#pragma unroll
    for (int tm = 0; tm < 4; ++tm) {
        const int lrow0 = wm * 64 + tm * 16;
#pragma unroll
        for (int tn = 0; tn < 2; ++tn) {
            const int col = bn + wn * 32 + tn * 16 + mrow;
            const float ac = alpha[col];
            const float bc = bias[col];
#pragma unroll
            for (int r = 0; r < 8; ++r) {
                const int lrow = lrow0 + r + half * 8;
                const int grow = bm + lrow;
                float val = acc[tm][tn][r] * (beta_in[grow] * beta_scale) * ac + bc;
                if (mode == 0) {
                    val = (val > 0.f) ? val : pw * val;
                    OutSign[(size_t)grow * NC + col] = sign_fp8(val);
                    float av = fabsf(val);
                    av += __shfl_xor(av, 1, 32);
                    av += __shfl_xor(av, 2, 32);
                    av += __shfl_xor(av, 4, 32);
                    av += __shfl_xor(av, 8, 32);
                    if (mrow == 0) atomicAdd(&rowsum[lrow], av);
                } else {
                    OutF32[(size_t)grow * NC + col] = val;
                }
            }
        }
    }

    if (mode == 0) {
        __syncthreads();
        for (int i = t; i < BM; i += 256)
            atomicAdd(&betasum_out[bm + i], rowsum[i]);
    }
}

extern "C" void kernel_launch(void* const* d_in, const int* in_sizes, int n_in,
                              void* d_out, int out_size, void* d_ws, size_t ws_size,
                              hipStream_t stream)
{
    (void)in_sizes; (void)n_in; (void)out_size; (void)ws_size;

    const float* x   = (const float*)d_in[0];
    const float* W1  = (const float*)d_in[1];
    const float* b1  = (const float*)d_in[2];
    const float* W2  = (const float*)d_in[3];
    const float* b2  = (const float*)d_in[4];
    const float* W3  = (const float*)d_in[5];
    const float* b3  = (const float*)d_in[6];
    const float* pw  = (const float*)d_in[7];
    float* out = (float*)d_out;

    char* p = (char*)d_ws;
    auto carve = [&](size_t bytes) {
        char* q = p;
        p += (bytes + 255) & ~(size_t)255;
        return q;
    };

    unsigned char* sX  = (unsigned char*)carve((size_t)N_SAMP * D_IN);
    unsigned char* sH1 = (unsigned char*)carve((size_t)N_SAMP * D_HID);
    unsigned char* sH2 = (unsigned char*)carve((size_t)N_SAMP * D_HID);
    unsigned char* sW1 = (unsigned char*)carve((size_t)D_HID * D_IN);
    unsigned char* sW2 = (unsigned char*)carve((size_t)D_HID * D_HID);
    unsigned char* sW3 = (unsigned char*)carve((size_t)D_OUT * D_HID);
    float* betaX = (float*)carve((size_t)N_SAMP * sizeof(float));
    float* bsum  = (float*)carve((size_t)2 * N_SAMP * sizeof(float));
    float* bsum1 = bsum;
    float* bsum2 = bsum + N_SAMP;
    float* a1 = (float*)carve((size_t)D_HID * sizeof(float));
    float* a2 = (float*)carve((size_t)D_HID * sizeof(float));
    float* a3 = (float*)carve((size_t)D_OUT * sizeof(float));

    zero_f32<<<(2 * N_SAMP + 255) / 256, 256, 0, stream>>>(bsum, 2 * N_SAMP);

    prep_sign_scale<<<N_SAMP, 256, 0, stream>>>(x,  sX,  betaX, D_IN);
    prep_sign_scale<<<D_HID,  256, 0, stream>>>(W1, sW1, a1,    D_IN);
    prep_sign_scale<<<D_HID,  256, 0, stream>>>(W2, sW2, a2,    D_HID);
    prep_sign_scale<<<D_OUT,  256, 0, stream>>>(W3, sW3, a3,    D_HID);

    dim3 blk(256);
    dim3 g12(N_SAMP / BM, D_HID / BN);
    xnor_wmma_gemm<<<g12, blk, 0, stream>>>(sX, sW1, betaX, 1.0f, a1, b1, pw,
                                            sH1, bsum1, nullptr,
                                            N_SAMP, D_HID, D_IN, 0);
    xnor_wmma_gemm<<<g12, blk, 0, stream>>>(sH1, sW2, bsum1, 1.0f / (float)D_HID,
                                            a2, b2, pw,
                                            sH2, bsum2, nullptr,
                                            N_SAMP, D_HID, D_HID, 0);
    dim3 g3(N_SAMP / BM, D_OUT / BN);
    xnor_wmma_gemm<<<g3, blk, 0, stream>>>(sH2, sW3, bsum2, 1.0f / (float)D_HID,
                                           a3, b3, nullptr,
                                           nullptr, nullptr, out,
                                           N_SAMP, D_OUT, D_HID, 1);
}
